// LTAE2d_52475910423215
// MI455X (gfx1250) — matrix operand order repaired and checked by reading, not compiled, not run on hardware
//
#include <hip/hip_runtime.h>
#include <cmath>

typedef __attribute__((ext_vector_type(16))) _Float16 v16h;
typedef __attribute__((ext_vector_type(8)))  _Float16 v8h;
typedef __attribute__((ext_vector_type(8)))  float    v8f;
typedef __attribute__((ext_vector_type(4)))  unsigned int v4u;
typedef __attribute__((ext_vector_type(8)))  int      v8i_;
typedef __attribute__((ext_vector_type(4)))  int      v4i_;

#define BB_  2
#define TT   20
#define CC   128
#define HH_  128
#define WW_  128
#define DM   256
#define NHD  16
#define MO   128
#define NS   16      // samples (w-positions) per workgroup
#define XRS  17      // f32 words per staged x row (16 data + 1 pad, TDM pad_enable)
#define XNROW 136    // padded f16 row length over c
#define OROW 264     // padded f16 row length over d (=256)
#define H1ROW 132

// ---- LDS region byte offsets (dynamic shared) ----
#define OFF_X     0            // xr f32 [TT*CC][XRS] = 174080 B ; later aliased as out16 [TT][NS][OROW] f16 (168960 B)
#define OFF_XN    174080       // [TT][NS][XNROW] f16 = 87040 B
#define OFF_SC    261120       // [NS*NHD][TT] f32 = 20480 B
#define OFF_PL    281600       // [NS][OROW] f16 = 8448 B
#define OFF_H1    290048       // [NS][H1ROW] f32 = 8448 B
#define OFF_ST    298496       // mu[256] + rstd[256] f32 = 2048 B
#define SMEM_BYTES 300544

#if defined(__has_builtin)
# if __has_builtin(__builtin_amdgcn_tensor_load_to_lds) && __has_builtin(__builtin_amdgcn_s_wait_tensorcnt)
#  define HAVE_TDM 1
# endif
#endif

__device__ __forceinline__ v8f wmma16(v16h a, v16h b, v8f c) {
  return __builtin_amdgcn_wmma_f32_16x16x32_f16(false, a, false, b, (short)0, c, false, false);
}

// A fragment (16x32 f16): lane<16 holds A[m][kb+0..7],A[m][kb+16..23];
// lane>=16 holds A[m][kb+8..15],A[m][kb+24..31]. m indexed externally via `row`.
template <typename P>
__device__ __forceinline__ v16h load_a_frag(P row, int kbase, int lane) {
  int koff = (lane >= 16) ? 8 : 0;
  v8h a = *(const v8h*)(row + kbase + koff);
  v8h b = *(const v8h*)(row + kbase + 16 + koff);
  return __builtin_shufflevector(a, b, 0,1,2,3,4,5,6,7,8,9,10,11,12,13,14,15);
}

// B fragment (32x16 f16, column n = lane%16, `colrow` = weight row for that column,
// row-major over K): lane<16 holds K=kbase+0..7,kbase+16..23; lane>=16 holds K=kbase+8..15,kbase+24..31.
__device__ __forceinline__ v16h load_b_frag(const _Float16* colrow, int kbase, int lane) {
  int koff = (lane >= 16) ? 8 : 0;
  v8h a = *(const v8h*)(colrow + kbase + koff);
  v8h b = *(const v8h*)(colrow + kbase + 16 + koff);
  return __builtin_shufflevector(a, b, 0,1,2,3,4,5,6,7,8,9,10,11,12,13,14,15);
}

// -------- prep: f16 weight conversion, Q-folded key weights, PE table --------
__global__ void ltae_prep(const float* __restrict__ inconv_w, const float* __restrict__ Q,
                          const float* __restrict__ k_w, const float* __restrict__ k_b,
                          const float* __restrict__ mlp_w, const int* __restrict__ bpos,
                          _Float16* __restrict__ wInconv, _Float16* __restrict__ wMLP,
                          _Float16* __restrict__ wQK, float* __restrict__ qkb,
                          float* __restrict__ pe)
{
  int i = blockIdx.x * blockDim.x + threadIdx.x;
  if (i < DM * CC) { wInconv[i] = (_Float16)inconv_w[i]; return; }
  i -= DM * CC;
  if (i < MO * DM) { wMLP[i] = (_Float16)mlp_w[i]; return; }
  i -= MO * DM;
  if (i < NHD * DM) {          // qk_w[h][c] = sum_d Q[h,d] * k_w[h*4+d][c]
    int h = i >> 8, c = i & 255;
    float s = 0.f;
    for (int d = 0; d < 4; ++d) s += Q[h * 4 + d] * k_w[(h * 4 + d) * DM + c];
    wQK[i] = (_Float16)s;
    return;
  }
  i -= NHD * DM;
  if (i < NHD) {
    float s = 0.f;
    for (int d = 0; d < 4; ++d) s += Q[i * 4 + d] * k_b[i * 4 + d];
    qkb[i] = s;
    return;
  }
  i -= NHD;
  if (i < BB_ * TT * DM) {     // pe[b][t][dd], dd pattern repeats every 16
    int dd = i & 255;
    int t  = (i >> 8) % TT;
    int b  = i / (256 * TT);
    int ii = dd & 15;
    float denom = powf(1000.0f, (2.0f * (float)(ii >> 1)) / 16.0f);
    float ang = (float)bpos[b * TT + t] / denom;
    pe[i] = (ii & 1) ? cosf(ang) : sinf(ang);
  }
}

// -------- main fused kernel: one workgroup = 16 samples (fixed b,h; w0..w0+15) --------
__global__ __launch_bounds__(256)
void ltae_main(const float* __restrict__ x,
               const float* __restrict__ innw, const float* __restrict__ innb,
               const float* __restrict__ inconv_b,
               const _Float16* __restrict__ wInconv, const _Float16* __restrict__ wQK,
               const float* __restrict__ qkb, const _Float16* __restrict__ wMLP,
               const float* __restrict__ mlpb,
               const float* __restrict__ lnw, const float* __restrict__ lnb,
               const float* __restrict__ onw, const float* __restrict__ onb,
               const float* __restrict__ pe, float* __restrict__ outp)
{
  extern __shared__ __align__(16) char smem[];
  float*    xr    = (float*)(smem + OFF_X);       // [t*CC+c][XRS]  (phase A-C)
  _Float16* out16 = (_Float16*)(smem + OFF_X);    // [t][n][OROW]   (phase D+, aliases xr)
  _Float16* xn16  = (_Float16*)(smem + OFF_XN);
  float*    scL   = (float*)(smem + OFF_SC);
  _Float16* plL   = (_Float16*)(smem + OFF_PL);
  float*    h1L   = (float*)(smem + OFF_H1);
  float*    muL   = (float*)(smem + OFF_ST);
  float*    rsL   = muL + 256;

  const int tid  = threadIdx.x;
  const int lane = tid & 31;
  const int wv   = tid >> 5;
  const int lm   = lane & 15;
  const int lhi  = lane >> 4;

  const int wg = blockIdx.x;
  const int w0 = (wg % (WW_ / NS)) * NS;
  const int hh = (wg / (WW_ / NS)) % HH_;
  const int bb = wg / ((WW_ / NS) * HH_);

  // warm caches for the weight fragments we'll need after the barrier
  __builtin_prefetch(wInconv + (wv * 16 + lm) * CC, 0, 1);
  __builtin_prefetch(wQK + lm * DM, 0, 1);

  // ---- Phase A: stage x tile into LDS as f32 [t*C+c][16 (+1 pad)] ----
  const float* xb = x + (size_t)bb * TT * CC * HH_ * WW_ + (size_t)hh * WW_ + w0;
#ifdef HAVE_TDM
  if (wv == 0) {
    // Tensor DMA: 2560 rows x 16 f32, row stride H*W elements, LDS pad 1 DWORD / 16 DWORDs
    unsigned long long ga = (unsigned long long)(uintptr_t)xb;
    unsigned int lds0 = (unsigned int)(uintptr_t)smem;   // LDS aperture: low 32 bits = LDS offset
    v4u g0;
    g0[0] = 1u;                                          // count = 1 valid descriptor
    g0[1] = lds0;                                        // lds_addr
    g0[2] = (unsigned int)ga;                            // global_addr[31:0]
    g0[3] = ((unsigned int)(ga >> 32) & 0x01FFFFFFu) | 0x80000000u;  // addr[56:32] | type=2
    v8i_ g1;
    g1[0] = (int)((2u << 16) | (1u << 20) | (3u << 22)); // data_size=4B, pad_en, interval=16dw, amount=1dw
    g1[1] = (int)(16u << 16);                            // tensor_dim0 = 16
    g1[2] = (int)(2560u << 16);                          // tensor_dim1 = 2560
    g1[3] = (int)(16u << 16);                            // tile_dim0 = 16
    g1[4] = (int)2560;                                   // tile_dim1 = 2560, tile_dim2 = 0
    g1[5] = (int)(HH_ * WW_);                            // tensor_dim0_stride = 16384 (lo32)
    g1[6] = 0;
    g1[7] = 0;
    v4i_ z4 = {0, 0, 0, 0};
#if __clang_major__ >= 23
    v8i_ z8 = {0, 0, 0, 0, 0, 0, 0, 0};
    __builtin_amdgcn_tensor_load_to_lds(g0, g1, z4, z4, z8, 0);
#else
    __builtin_amdgcn_tensor_load_to_lds(g0, g1, z4, z4, 0);
#endif
    __builtin_amdgcn_s_wait_tensorcnt(0);
  }
#else
  for (int e = tid; e < TT * CC * NS; e += 256) {
    int i = e & 15;
    int rc = e >> 4;                    // t*CC + c
    const float* gp = xb + (size_t)rc * (HH_ * WW_) + i;
    xr[rc * XRS + i] = *gp;
  }
#endif
  __syncthreads();

  // ---- Phase B: GroupNorm stats, one thread per (sample n, group g) ----
  {
    int n = tid & 15, g = tid >> 4;
    float s = 0.f, ss = 0.f;
    for (int t = 0; t < TT; ++t) {
      const float* row = &xr[(t * CC + g * 8) * XRS + n];
      #pragma unroll
      for (int j = 0; j < 8; ++j) { float v = row[j * XRS]; s += v; ss += v * v; }
    }
    float mu  = s * (1.0f / 160.0f);
    float var = ss * (1.0f / 160.0f) - mu * mu;
    muL[n * NHD + g] = mu;
    rsL[n * NHD + g] = rsqrtf(var + 1e-5f);
  }
  __syncthreads();

  // ---- Phase C: normalize + transpose -> f16 xn[t][n][c] ----
  for (int e = tid; e < TT * CC * NS; e += 256) {
    int i = e & 15;
    int c = (e >> 4) & 127;
    int t = e >> 11;
    int g = c >> 3;
    float v  = xr[(t * CC + c) * XRS + i];
    float nv = (v - muL[i * NHD + g]) * rsL[i * NHD + g] * innw[c] + innb[c];
    xn16[(t * NS + i) * XNROW + c] = (_Float16)nv;
  }
  __syncthreads();   // xr dead; region becomes out16

  // ---- Phase D: inconv GEMM (16x128)@(128->256) per t, WMMA f16->f32, +bias+PE ----
  for (int ntl = 0; ntl < 2; ++ntl) {
    int nt   = wv + ntl * 8;
    int dcol = nt * 16 + lm;
    const _Float16* wrow = wInconv + dcol * CC;
    v16h Bf[4];
    #pragma unroll
    for (int k = 0; k < 4; ++k) Bf[k] = load_b_frag(wrow, k * 32, lane);
    float bias = inconv_b[dcol];
    const float* perow = pe + (size_t)bb * TT * DM + dcol;
    for (int t = 0; t < TT; ++t) {
      v8f D = {0.f, 0.f, 0.f, 0.f, 0.f, 0.f, 0.f, 0.f};
      const _Float16* arow = &xn16[(t * NS + lm) * XNROW];
      #pragma unroll
      for (int k = 0; k < 4; ++k) D = wmma16(load_a_frag(arow, k * 32, lane), Bf[k], D);
      float add = bias + perow[t * DM];
      _Float16* orow = &out16[(t * NS + lhi * 8) * OROW + dcol];
      #pragma unroll
      for (int r = 0; r < 8; ++r) orow[r * OROW] = (_Float16)(D[r] + add);
    }
  }
  __syncthreads();

  // ---- Phase E: folded key-projection scores (16 samples x 16 heads) per t ----
  {
    const _Float16* qrow = wQK + lm * DM;
    v16h Bf[8];
    #pragma unroll
    for (int k = 0; k < 8; ++k) Bf[k] = load_b_frag(qrow, k * 32, lane);
    float qb = qkb[lm];
    for (int t = wv; t < TT; t += 8) {
      v8f D = {0.f, 0.f, 0.f, 0.f, 0.f, 0.f, 0.f, 0.f};
      const _Float16* arow = &out16[(t * NS + lm) * OROW];
      #pragma unroll
      for (int k = 0; k < 8; ++k) D = wmma16(load_a_frag(arow, k * 32, lane), Bf[k], D);
      #pragma unroll
      for (int r = 0; r < 8; ++r) {
        int n = r + 8 * lhi;
        scL[(n * NHD + lm) * TT + t] = (D[r] + qb) * 0.5f;  // 1/sqrt(d_k)
      }
    }
  }
  __syncthreads();

  // ---- Phase F+G: softmax over t and attention pooling, thread = (n,h) ----
  {
    float* srow = &scL[tid * TT];         // tid = n*16 + h
    int n = tid >> 4, hd = tid & 15;
    float mx = -1e30f;
    for (int t = 0; t < TT; ++t) mx = fmaxf(mx, srow[t]);
    float sum = 0.f;
    for (int t = 0; t < TT; ++t) { float e = __expf(srow[t] - mx); srow[t] = e; sum += e; }
    float inv = 1.0f / sum;
    float acc[16];
    #pragma unroll
    for (int e2 = 0; e2 < 16; ++e2) acc[e2] = 0.f;
    for (int t = 0; t < TT; ++t) {
      float a = srow[t] * inv;
      const _Float16* vrow = &out16[(t * NS + n) * OROW + hd * 16];
      #pragma unroll
      for (int e2 = 0; e2 < 16; ++e2) acc[e2] += a * (float)vrow[e2];
    }
    _Float16* prow = &plL[n * OROW + hd * 16];
    #pragma unroll
    for (int e2 = 0; e2 < 16; ++e2) prow[e2] = (_Float16)acc[e2];
  }
  __syncthreads();

  // ---- Phase H: MLP GEMM (16x256)@(256->128), one N-tile per wave ----
  {
    int ocol = wv * 16 + lm;
    const _Float16* mrow = wMLP + ocol * DM;
    v16h Bf[8];
    #pragma unroll
    for (int k = 0; k < 8; ++k) Bf[k] = load_b_frag(mrow, k * 32, lane);
    v8f D = {0.f, 0.f, 0.f, 0.f, 0.f, 0.f, 0.f, 0.f};
    const _Float16* arow = &plL[lm * OROW];
    #pragma unroll
    for (int k = 0; k < 8; ++k) D = wmma16(load_a_frag(arow, k * 32, lane), Bf[k], D);
    float bias = mlpb[ocol];
    #pragma unroll
    for (int r = 0; r < 8; ++r) h1L[(r + 8 * lhi) * H1ROW + ocol] = D[r] + bias;
  }
  __syncthreads();

  // ---- Phase I: LayerNorm -> exact GELU -> out GroupNorm (thread per sample) ----
  if (tid < NS) {
    float* row = &h1L[tid * H1ROW];
    float s = 0.f, ss = 0.f;
    for (int o = 0; o < MO; ++o) { float v = row[o]; s += v; ss += v * v; }
    float mu  = s * (1.0f / MO);
    float var = ss * (1.0f / MO) - mu * mu;
    float rs  = rsqrtf(var + 1e-5f);
    for (int o = 0; o < MO; ++o) {
      float v = (row[o] - mu) * rs * lnw[o] + lnb[o];
      row[o] = 0.5f * v * (1.0f + erff(v * 0.70710678118f));
    }
    for (int g = 0; g < NHD; ++g) {
      float s2 = 0.f, ss2 = 0.f;
      #pragma unroll
      for (int j = 0; j < 8; ++j) { float v = row[g * 8 + j]; s2 += v; ss2 += v * v; }
      float mu2  = s2 * 0.125f;
      float var2 = ss2 * 0.125f - mu2 * mu2;
      float rs2  = rsqrtf(var2 + 1e-5f);
      #pragma unroll
      for (int j = 0; j < 8; ++j) {
        int o = g * 8 + j;
        row[o] = (row[o] - mu2) * rs2 * onw[o] + onb[o];
      }
    }
  }
  __syncthreads();

  // ---- Output write: out[b][co][h][w0+n], coalesced over n ----
  {
    int nl = tid & 15;
    int cr = tid >> 4;
    #pragma unroll
    for (int j = 0; j < 8; ++j) {
      int co = cr + j * 16;
      outp[(((size_t)bb * MO + co) * HH_ + hh) * WW_ + w0 + nl] = h1L[nl * H1ROW + co];
    }
  }
}

extern "C" void kernel_launch(void* const* d_in, const int* in_sizes, int n_in,
                              void* d_out, int out_size, void* d_ws, size_t ws_size,
                              hipStream_t stream) {
  const float* x        = (const float*)d_in[0];
  const int*   bpos     = (const int*)d_in[1];
  const float* innw     = (const float*)d_in[2];
  const float* innb     = (const float*)d_in[3];
  const float* inconv_w = (const float*)d_in[4];
  const float* inconv_b = (const float*)d_in[5];
  const float* Q        = (const float*)d_in[6];
  const float* k_w      = (const float*)d_in[7];
  const float* k_b      = (const float*)d_in[8];
  const float* mlp_w    = (const float*)d_in[9];
  const float* mlp_b    = (const float*)d_in[10];
  const float* ln_w     = (const float*)d_in[11];
  const float* ln_b     = (const float*)d_in[12];
  const float* on_w     = (const float*)d_in[13];
  const float* on_b     = (const float*)d_in[14];
  float* outp = (float*)d_out;

  char* ws = (char*)d_ws;
  _Float16* wInconv = (_Float16*)(ws + 0);        // 256*128 f16
  _Float16* wMLP    = (_Float16*)(ws + 65536);    // 128*256 f16
  _Float16* wQK     = (_Float16*)(ws + 131072);   // 16*256 f16
  float*    qkb     = (float*)(ws + 139264);      // 16 f32
  float*    pe      = (float*)(ws + 139328);      // 2*20*256 f32

  int prepN = DM * CC + MO * DM + NHD * DM + NHD + BB_ * TT * DM;
  ltae_prep<<<(prepN + 255) / 256, 256, 0, stream>>>(inconv_w, Q, k_w, k_b, mlp_w, bpos,
                                                     wInconv, wMLP, wQK, qkb, pe);

  hipFuncSetAttribute((const void*)ltae_main,
                      hipFuncAttributeMaxDynamicSharedMemorySize, SMEM_BYTES);
  dim3 grid(BB_ * HH_ * (WW_ / NS));   // 2048 workgroups of 256 threads (8 waves)
  ltae_main<<<grid, 256, SMEM_BYTES, stream>>>(x, innw, innb, inconv_b, wInconv, wQK, qkb,
                                               wMLP, mlp_b, ln_w, ln_b, on_w, on_b, pe, outp);
}
